// RelativeAttention_13864154432323
// MI455X (gfx1250) — matrix operand order repaired and checked by reading, not compiled, not run on hardware
//
#include <hip/hip_runtime.h>
#include <hip/hip_bf16.h>
#include <cstdint>

// ---------------------------------------------------------------------------
// Relative-position multi-head attention for MI455X (gfx1250, wave32, WMMA).
//   B=16, N=577 tokens, C=768, 12 heads x 64.  ~60 GFLOP, all data L2-resident.
//   bf16 WMMA (f32 accumulate) everywhere; flash-style streaming softmax so the
//   255MB attention matrix is never materialized.
// ---------------------------------------------------------------------------

#define B_SZ   16
#define NTOK   577
#define NT     37          // ceil(577/16) row tiles
#define CDIM   768
#define C3     2304
#define NHEAD  12
#define HD     64
#define NPAD   592         // padded key dim for vT rows (multiple of 16)

typedef __bf16 bf16;
typedef bf16  v16bf __attribute__((ext_vector_type(16)));
typedef float v8f   __attribute__((ext_vector_type(8)));
typedef unsigned int u32;
typedef u32 u32x4 __attribute__((ext_vector_type(4)));

// exact parameter types of the async-to-LDS builtin:
//   (int __vector(4) addrspace(1)*, int __vector(4) addrspace(3)*, imm, imm)
typedef int v4i __attribute__((vector_size(16)));
typedef __attribute__((address_space(1))) v4i gv4i;   // global
typedef __attribute__((address_space(3))) v4i lv4i;   // LDS

union FragU { v16bf v; u32x4 q[2]; };

#if __has_builtin(__builtin_amdgcn_global_load_async_to_lds_b128)
#define HAVE_ASYNC 1
#endif

// 16-byte global -> LDS copy (async on CDNA5 when available; tracked by ASYNCcnt)
__device__ __forceinline__ void copy16_to_lds(const bf16* g, bf16* l) {
#ifdef HAVE_ASYNC
  __builtin_amdgcn_global_load_async_to_lds_b128((gv4i*)(g), (lv4i*)(l), 0, 0);
#else
  *reinterpret_cast<u32x4*>(l) = *reinterpret_cast<const u32x4*>(g);
#endif
}

__device__ __forceinline__ void wait_async_all() {
#ifdef HAVE_ASYNC
#if __has_builtin(__builtin_amdgcn_s_wait_asynccnt)
  __builtin_amdgcn_s_wait_asynccnt(0);
#else
  asm volatile("s_wait_asynccnt 0" ::: "memory");
#endif
#endif
}

// A fragment (16x32 bf16, M x K): lane = 16*half + m; 16 contiguous K values
// starting at 16*half.  `base` points at row m, column kk.
__device__ __forceinline__ v16bf load_frag_a(const bf16* base, int kq) {
  FragU f;
  f.q[0] = *reinterpret_cast<const u32x4*>(base + 2 * kq);
  f.q[1] = *reinterpret_cast<const u32x4*>(base + 2 * kq + 8);
  return f.v;
}

// B fragment (32x16 bf16, K x N): lane = 16*half + n; 16 contiguous K values
// starting at 16*half.  `p` points at column n, row 16*half.
__device__ __forceinline__ v16bf load_frag_b(const bf16* p) {
  FragU f;
  f.q[0] = *reinterpret_cast<const u32x4*>(p);
  f.q[1] = *reinterpret_cast<const u32x4*>(p + 8);
  return f.v;
}

__device__ __forceinline__ v8f wmma_bf(v16bf a, v16bf b, v8f c) {
  // (neg_a, A, neg_b, B, c_mod, C, reuse_a, reuse_b)
  return __builtin_amdgcn_wmma_f32_16x16x32_bf16(false, a, false, b, (short)0, c,
                                                 false, false);
}

// ---------------------------------------------------------------------------
// Elementwise f32 -> bf16 cast
// ---------------------------------------------------------------------------
__global__ void cast_kernel(const float* __restrict__ src, bf16* __restrict__ dst,
                            int n) {
  int i = blockIdx.x * blockDim.x + threadIdx.x;
  if (i < n) dst[i] = (bf16)src[i];
}

// ---------------------------------------------------------------------------
// bias_full[h][n][m] = bias_table[rel_idx[n][m]][h]   (f32, reused by all batches)
// ---------------------------------------------------------------------------
__global__ void bias_kernel(const float* __restrict__ table,
                            const int* __restrict__ rel,
                            float* __restrict__ out) {
  int i = blockIdx.x * blockDim.x + threadIdx.x;
  if (i >= NTOK * NTOK) return;
  int idx = rel[i];
#pragma unroll
  for (int h = 0; h < NHEAD; ++h)
    out[(size_t)h * NTOK * NTOK + i] = table[idx * NHEAD + h];
}

// ---------------------------------------------------------------------------
// QKV GEMM: y[n, 0:2304] = x[n,:] @ qkv_w^T   (per batch)
//   block = 8 waves; 16x768 x-tile staged to LDS via async copies;
//   each wave owns a 16x64 output tile (4 accs, A-frag reused 4x).
//   Epilogue writes q (scaled), k head-major (B,H,N,64) and v transposed
//   (B,H,64,NPAD) in bf16.
// ---------------------------------------------------------------------------
__launch_bounds__(256)
__global__ void qkv_gemm(const bf16* __restrict__ xb, const bf16* __restrict__ wq,
                         bf16* __restrict__ qo, bf16* __restrict__ ko,
                         bf16* __restrict__ vo) {
  __shared__ bf16 lds_a[16 * 776];              // 16 rows, +8 pad
  int b = blockIdx.x / NT, rt = blockIdx.x % NT;
  int n0 = rt * 16;
  const bf16* xrow = xb + (size_t)b * NTOK * CDIM;

  for (int c = threadIdx.x; c < 16 * 96; c += 256) {   // 96 16B-chunks per row
    int row = c / 96, ch = c % 96;
    int tok = n0 + row; if (tok > NTOK - 1) tok = NTOK - 1;
    copy16_to_lds(xrow + (size_t)tok * CDIM + ch * 8, lds_a + row * 776 + ch * 8);
  }
  wait_async_all();
  __syncthreads();

  int wave = threadIdx.x >> 5, lane = threadIdx.x & 31;
  int ct = blockIdx.y * 8 + wave;
  if (ct >= 36) return;                          // 2304/64 = 36 column tiles
  int j0 = ct * 64;
  int m = lane & 15, half = lane >> 4;
  int kq = half * 8;
  const bf16* arow = lds_a + m * 776;

  v8f acc[4] = {{}, {}, {}, {}};
#pragma unroll 4
  for (int kk = 0; kk < CDIM; kk += 32) {
    v16bf af = load_frag_a(arow + kk, kq);
#pragma unroll
    for (int a = 0; a < 4; ++a) {
      const bf16* bp = wq + (size_t)(j0 + a * 16 + m) * CDIM + kk + half * 16;
      acc[a] = wmma_bf(af, load_frag_b(bp), acc[a]);
    }
  }

#pragma unroll
  for (int a = 0; a < 4; ++a) {
    int j = j0 + a * 16 + m;                     // output column (n of C tile)
    int which = j / CDIM;                        // 0=q 1=k 2=v
    int rem = j % CDIM;
    int h = rem >> 6, dd = rem & 63;
    size_t bh = (size_t)(b * NHEAD + h);
#pragma unroll
    for (int r = 0; r < 8; ++r) {
      int tok = n0 + r + 8 * half;
      if (tok >= NTOK) continue;
      float val = acc[a][r];
      if (which == 0)      qo[(bh * NTOK + tok) * HD + dd] = (bf16)(val * 0.125f);
      else if (which == 1) ko[(bh * NTOK + tok) * HD + dd] = (bf16)val;
      else                 vo[(bh * HD + dd) * NPAD + tok] = (bf16)val;
    }
  }
}

// ---------------------------------------------------------------------------
// Flash attention: one wave per (b, h, 16-row q tile).  Streams keys in tiles
// of 64; online softmax with per-row stats kept in the (r, half) slots that
// match the WMMA C-layout; P bounced through LDS (C-layout -> A-layout, bf16).
// ---------------------------------------------------------------------------
__launch_bounds__(128)
__global__ void attn_kernel(const bf16* __restrict__ q, const bf16* __restrict__ k,
                            const bf16* __restrict__ vT,
                            const float* __restrict__ bias,
                            bf16* __restrict__ ctx) {
  __shared__ bf16 ldsP[4][16 * 72];              // per-wave P tile, stride 72
  int bh = blockIdx.x;                           // b*12 + h
  int b = bh / NHEAD, h = bh % NHEAD;
  int wave = threadIdx.x >> 5, lane = threadIdx.x & 31;
  int rt = blockIdx.y * 4 + wave;
  if (rt >= NT) return;
  int n0 = rt * 16;
  int m_ = lane & 15, half = lane >> 4;
  int kq = half * 8;

  const bf16* qbase = q  + (size_t)bh * NTOK * HD;
  const bf16* kbase = k  + (size_t)bh * NTOK * HD;
  const bf16* vbase = vT + (size_t)bh * HD * NPAD;
  const float* bbase = bias + (size_t)h * NTOK * NTOK;
  bf16* P = &ldsP[wave][0];

  // q fragments for this row tile stay in registers for all 10 key tiles
  int qtok = n0 + m_; if (qtok > NTOK - 1) qtok = NTOK - 1;
  v16bf qf0 = load_frag_a(qbase + (size_t)qtok * HD, kq);
  v16bf qf1 = load_frag_a(qbase + (size_t)qtok * HD + 32, kq);

  v8f O[4] = {{}, {}, {}, {}};
  float runmax[8], runsum[8];
#pragma unroll
  for (int r = 0; r < 8; ++r) { runmax[r] = -3e38f; runsum[r] = 0.f; }

  for (int m0 = 0; m0 < 640; m0 += 64) {         // 10 key tiles (577 -> 640)
    // ---- S = q . k^T  (16 x 64) ----
    v8f S[4] = {{}, {}, {}, {}};
#pragma unroll
    for (int a = 0; a < 4; ++a) {
      int key = m0 + a * 16 + m_;
      int keyc = key > NTOK - 1 ? NTOK - 1 : key;
      const bf16* kp = kbase + (size_t)keyc * HD + half * 16;
      S[a] = wmma_bf(qf0, load_frag_b(kp), S[a]);
      S[a] = wmma_bf(qf1, load_frag_b(kp + 32), S[a]);
    }
    // ---- bias + key mask ----
#pragma unroll
    for (int a = 0; a < 4; ++a) {
      int key = m0 + a * 16 + m_;
      int keyc = key > NTOK - 1 ? NTOK - 1 : key;
#pragma unroll
      for (int r = 0; r < 8; ++r) {
        int tok = n0 + r + 8 * half;
        int tokc = tok > NTOK - 1 ? NTOK - 1 : tok;
        float s = S[a][r] + bbase[(size_t)tokc * NTOK + keyc];
        S[a][r] = (key >= NTOK) ? -1e30f : s;
      }
    }
    // ---- online softmax: row stats live in (r, half) slots of the half-wave
    float mnew[8], alpha[8], psum[8];
#pragma unroll
    for (int r = 0; r < 8; ++r) {
      float mx = fmaxf(fmaxf(S[0][r], S[1][r]), fmaxf(S[2][r], S[3][r]));
      mx = fmaxf(mx, __shfl_xor(mx, 1, 32));
      mx = fmaxf(mx, __shfl_xor(mx, 2, 32));
      mx = fmaxf(mx, __shfl_xor(mx, 4, 32));
      mx = fmaxf(mx, __shfl_xor(mx, 8, 32));
      mnew[r]  = fmaxf(runmax[r], mx);
      alpha[r] = __expf(runmax[r] - mnew[r]);
      runmax[r] = mnew[r];
      psum[r] = 0.f;
    }
    // ---- P = exp(S - mnew), C-layout -> LDS (bf16, A-layout readable) ----
#pragma unroll
    for (int a = 0; a < 4; ++a) {
      int keyl = a * 16 + m_;
#pragma unroll
      for (int r = 0; r < 8; ++r) {
        float p = __expf(S[a][r] - mnew[r]);
        psum[r] += p;
        P[(r + 8 * half) * 72 + keyl] = (bf16)p;
      }
    }
#pragma unroll
    for (int r = 0; r < 8; ++r) {
      float s = psum[r];
      s += __shfl_xor(s, 1, 32); s += __shfl_xor(s, 2, 32);
      s += __shfl_xor(s, 4, 32); s += __shfl_xor(s, 8, 32);
      runsum[r] = runsum[r] * alpha[r] + s;
    }
#pragma unroll
    for (int a = 0; a < 4; ++a)
#pragma unroll
      for (int r = 0; r < 8; ++r) O[a][r] *= alpha[r];

    // ---- O += P @ V  (A = P from LDS, B = vT rows, contiguous in keys) ----
    v16bf p0 = load_frag_a(P + m_ * 72, kq);
    v16bf p1 = load_frag_a(P + m_ * 72 + 32, kq);
#pragma unroll
    for (int a = 0; a < 4; ++a) {
      const bf16* vp = vbase + (size_t)(a * 16 + m_) * NPAD + m0 + half * 16;
      O[a] = wmma_bf(p0, load_frag_b(vp), O[a]);
      O[a] = wmma_bf(p1, load_frag_b(vp + 32), O[a]);
    }
  }

  // ---- normalize and store context (B, N, H*64) bf16, coalesced in dd ----
#pragma unroll
  for (int a = 0; a < 4; ++a) {
    int dd = a * 16 + m_;
#pragma unroll
    for (int r = 0; r < 8; ++r) {
      int tok = n0 + r + 8 * half;
      if (tok < NTOK)
        ctx[((size_t)b * NTOK + tok) * CDIM + h * HD + dd] =
            (bf16)(O[a][r] / runsum[r]);
    }
  }
}

// ---------------------------------------------------------------------------
// Output projection: out[n,:] = ctx[n,:] @ proj_w^T + proj_b   (f32 out)
// ---------------------------------------------------------------------------
__launch_bounds__(128)
__global__ void proj_gemm(const bf16* __restrict__ ctx, const bf16* __restrict__ wp,
                          const float* __restrict__ pb, float* __restrict__ out) {
  __shared__ bf16 lds_a[16 * 776];
  int b = blockIdx.x / NT, rt = blockIdx.x % NT;
  int n0 = rt * 16;
  const bf16* crow = ctx + (size_t)b * NTOK * CDIM;

  for (int c = threadIdx.x; c < 16 * 96; c += 128) {
    int row = c / 96, ch = c % 96;
    int tok = n0 + row; if (tok > NTOK - 1) tok = NTOK - 1;
    copy16_to_lds(crow + (size_t)tok * CDIM + ch * 8, lds_a + row * 776 + ch * 8);
  }
  wait_async_all();
  __syncthreads();

  int wave = threadIdx.x >> 5, lane = threadIdx.x & 31;
  int ct = blockIdx.y * 4 + wave;               // 0..11 (768/64), exact
  int j0 = ct * 64;
  int m = lane & 15, half = lane >> 4;
  int kq = half * 8;
  const bf16* arow = lds_a + m * 776;

  v8f acc[4] = {{}, {}, {}, {}};
#pragma unroll 4
  for (int kk = 0; kk < CDIM; kk += 32) {
    v16bf af = load_frag_a(arow + kk, kq);
#pragma unroll
    for (int a = 0; a < 4; ++a) {
      const bf16* bp = wp + (size_t)(j0 + a * 16 + m) * CDIM + kk + half * 16;
      acc[a] = wmma_bf(af, load_frag_b(bp), acc[a]);
    }
  }

#pragma unroll
  for (int a = 0; a < 4; ++a) {
    int j = j0 + a * 16 + m;
    float bj = pb[j];
#pragma unroll
    for (int r = 0; r < 8; ++r) {
      int tok = n0 + r + 8 * half;
      if (tok < NTOK)
        out[((size_t)b * NTOK + tok) * CDIM + j] = acc[a][r] + bj;
    }
  }
}

// ---------------------------------------------------------------------------
extern "C" void kernel_launch(void* const* d_in, const int* in_sizes, int n_in,
                              void* d_out, int out_size, void* d_ws, size_t ws_size,
                              hipStream_t stream) {
  const float* x          = (const float*)d_in[0];
  const float* qkv_w      = (const float*)d_in[1];
  const float* proj_w     = (const float*)d_in[2];
  const float* proj_b     = (const float*)d_in[3];
  const float* bias_table = (const float*)d_in[4];
  const int*   rel_idx    = (const int*)d_in[5];
  float* out = (float*)d_out;
  (void)in_sizes; (void)n_in; (void)out_size; (void)ws_size;

  char* base = (char*)d_ws;
  size_t off = 0;
  auto alloc = [&](size_t bytes) -> void* {
    void* p = base + off;
    off += (bytes + 255) & ~(size_t)255;
    return p;
  };
  bf16*  xb    = (bf16*) alloc((size_t)B_SZ * NTOK * CDIM * 2);   // 14.2 MB
  bf16*  wqkv  = (bf16*) alloc((size_t)C3 * CDIM * 2);            //  3.5 MB
  bf16*  wproj = (bf16*) alloc((size_t)CDIM * CDIM * 2);          //  1.2 MB
  bf16*  qb    = (bf16*) alloc((size_t)B_SZ * NHEAD * NTOK * HD * 2);
  bf16*  kb    = (bf16*) alloc((size_t)B_SZ * NHEAD * NTOK * HD * 2);
  bf16*  vT    = (bf16*) alloc((size_t)B_SZ * NHEAD * HD * NPAD * 2);
  float* biasf = (float*)alloc((size_t)NHEAD * NTOK * NTOK * 4);  // 16.0 MB
  bf16*  ctx   = (bf16*) alloc((size_t)B_SZ * NTOK * CDIM * 2);   // ~92 MB total

  int nx = B_SZ * NTOK * CDIM;
  cast_kernel<<<(nx + 255) / 256, 256, 0, stream>>>(x, xb, nx);
  int nw1 = C3 * CDIM;
  cast_kernel<<<(nw1 + 255) / 256, 256, 0, stream>>>(qkv_w, wqkv, nw1);
  int nw2 = CDIM * CDIM;
  cast_kernel<<<(nw2 + 255) / 256, 256, 0, stream>>>(proj_w, wproj, nw2);

  (void)hipMemsetAsync(vT, 0, (size_t)B_SZ * NHEAD * HD * NPAD * 2, stream);
  bias_kernel<<<(NTOK * NTOK + 255) / 256, 256, 0, stream>>>(bias_table, rel_idx,
                                                             biasf);

  qkv_gemm<<<dim3(B_SZ * NT, 5), 256, 0, stream>>>(xb, wqkv, qb, kb, vT);
  attn_kernel<<<dim3(B_SZ * NHEAD, 10), 128, 0, stream>>>(qb, kb, vT, biasf, ctx);
  proj_gemm<<<dim3(B_SZ * NT, 3), 128, 0, stream>>>(ctx, wproj, proj_b, out);
}
